// Attention_11347303596474
// MI455X (gfx1250) — matrix operand order repaired and checked by reading, not compiled, not run on hardware
//
#include <hip/hip_runtime.h>
#include <hip/hip_bf16.h>

// ---------------------------------------------------------------------------
// MI455X (gfx1250) Perceiver cross-attention.
//   B=4 N=4096 M=512 DIM=1024 H=8 Dh=64 INNER=512
//   ~56 GFLOP, ~90MB traffic. All matmuls via v_wmma_f32_16x16x32_bf16.
//   GEMMs register-blocked 32x64 per wave: 8 independent accumulators per
//   k-step so WMMAs pipeline instead of chaining D->C RAW hazards.
//   Attention K/V staged with global_load_async_to_lds_b128 (ASYNCcnt).
//   __launch_bounds__(.,1) lifts the VGPR budget to kill epilogue spills.
// ---------------------------------------------------------------------------

typedef __bf16 bf16;
typedef __attribute__((ext_vector_type(16))) __bf16 v16bf;
typedef __attribute__((ext_vector_type(8)))  __bf16 v8bf;
typedef __attribute__((ext_vector_type(8)))  float  v8f;
typedef __attribute__((ext_vector_type(4)))  int    v4i;

#define GLOBAL_AS __attribute__((address_space(1)))
#define LDS_AS    __attribute__((address_space(3)))

constexpr int kDIM   = 1024;
constexpr int kHEADS = 8;
constexpr int kDH    = 64;
constexpr int kINNER = 512;   // HEADS * DH
constexpr int kB     = 4;
constexpr int kN     = 4096;
constexpr int kM     = 512;
constexpr float kSCALE = 0.125f;  // 64^-0.5

#if defined(__has_builtin)
#if __has_builtin(__builtin_amdgcn_global_load_async_to_lds_b128)
#define HAVE_ASYNC_LDS 1
#endif
#endif

static __device__ __forceinline__ v8f wmma_bf16(v16bf a, v16bf b, v8f c) {
  return __builtin_amdgcn_wmma_f32_16x16x32_bf16(false, a, false, b,
                                                 (short)0, c, false, false);
}

// -------------------- weight transpose + f32 -> bf16 ----------------------
__global__ void k_cvt_t(const float* __restrict__ W, bf16* __restrict__ Wt,
                        int K, int Nc) {
  int idx = blockIdx.x * blockDim.x + threadIdx.x;
  if (idx >= K * Nc) return;
  int k = idx / Nc;
  int n = idx - k * Nc;
  Wt[(size_t)n * K + k] = (bf16)W[idx];
}

// -------------------- LayerNorm (one wave per 1024-wide row) ---------------
__global__ void k_ln(const float* __restrict__ X, const float* __restrict__ g,
                     const float* __restrict__ bta, bf16* __restrict__ Y,
                     int rows) {
  int wid  = (blockIdx.x * blockDim.x + threadIdx.x) >> 5;
  int lane = threadIdx.x & 31;
  if (wid >= rows) return;
  const float* xr = X + (size_t)wid * kDIM;
  float v[32];
  float s = 0.f;
#pragma unroll
  for (int i = 0; i < 32; ++i) { v[i] = xr[lane + 32 * i]; s += v[i]; }
#pragma unroll
  for (int m = 16; m >= 1; m >>= 1) s += __shfl_xor(s, m, 32);
  float mu = s * (1.0f / kDIM);
  float q = 0.f;
#pragma unroll
  for (int i = 0; i < 32; ++i) { float d = v[i] - mu; q += d * d; }
#pragma unroll
  for (int m = 16; m >= 1; m >>= 1) q += __shfl_xor(q, m, 32);
  float rstd = rsqrtf(q * (1.0f / kDIM) + 1e-5f);
  bf16* yr = Y + (size_t)wid * kDIM;
#pragma unroll
  for (int i = 0; i < 32; ++i) {
    int c = lane + 32 * i;
    yr[c] = (bf16)((v[i] - mu) * rstd * g[c] + bta[c]);
  }
}

// ---- fragment loaders (ISA 7.12.2 layouts) --------------------------------
static __device__ __forceinline__ v16bf load_a_frag(const bf16* row, int kb,
                                                    int hf) {
  // A 16x32: e0..7 -> K = kb + hf*8 + e ; e8..15 -> K = kb + 16 + hf*8 + e
  v8bf a0 = *(const v8bf*)(row + kb + hf * 8);
  v8bf a1 = *(const v8bf*)(row + kb + 16 + hf * 8);
  v16bf av;
#pragma unroll
  for (int t = 0; t < 8; ++t) { av[t] = a0[t]; av[t + 8] = a1[t]; }
  return av;
}
static __device__ __forceinline__ v16bf load_b_frag(const bf16* col, int kb,
                                                    int hf) {
  // B 32x16: e0..7 -> K = kb + hf*8 + e ; e8..15 -> K = kb + 16 + hf*8 + e
  v8bf b0 = *(const v8bf*)(col + kb + hf * 8);
  v8bf b1 = *(const v8bf*)(col + kb + 16 + hf * 8);
  v16bf bv;
#pragma unroll
  for (int t = 0; t < 8; ++t) { bv[t] = b0[t]; bv[t + 8] = b1[t]; }
  return bv;
}

// -------------------- bf16 GEMM: C = scale * (A @ Bt^T) --------------------
// One wave per 32x64 output tile: 2 A-frags x 4 B-frags -> 8 independent
// WMMAs per 32-deep k-step (no D->C serialization).
__global__ void __launch_bounds__(128, 1)
k_gemm_bf16(const bf16* __restrict__ A,
            const bf16* __restrict__ Bt,
            bf16* __restrict__ C,
            int cols, int kd, float scale) {
  int wave = (blockIdx.x * blockDim.x + threadIdx.x) >> 5;
  int lane = threadIdx.x & 31;
  int l16 = lane & 15, hf = lane >> 4;
  int tn = cols >> 6;
  int m0 = (wave / tn) << 5;
  int n0 = (wave % tn) << 6;
  const bf16* ar[2];
  ar[0] = A + (size_t)(m0 + l16) * kd;
  ar[1] = ar[0] + (size_t)16 * kd;
  const bf16* bc = Bt + (size_t)(n0 + l16) * kd;
  v8f acc[2][4];
#pragma unroll
  for (int i = 0; i < 2; ++i)
#pragma unroll
    for (int j = 0; j < 4; ++j) acc[i][j] = (v8f){};

  for (int kb = 0; kb < kd; kb += 32) {
    v16bf av[2], bv[4];
#pragma unroll
    for (int i = 0; i < 2; ++i) av[i] = load_a_frag(ar[i], kb, hf);
#pragma unroll
    for (int j = 0; j < 4; ++j) bv[j] = load_b_frag(bc + (size_t)(j * 16) * kd, kb, hf);
#pragma unroll
    for (int i = 0; i < 2; ++i)
#pragma unroll
      for (int j = 0; j < 4; ++j)
        acc[i][j] = wmma_bf16(av[i], bv[j], acc[i][j]);
  }
#pragma unroll
  for (int i = 0; i < 2; ++i)
#pragma unroll
    for (int j = 0; j < 4; ++j) {
      bf16* crow = C + (size_t)(m0 + i * 16 + hf * 8) * cols + n0 + j * 16 + l16;
#pragma unroll
      for (int r = 0; r < 8; ++r)
        crow[(size_t)r * cols] = (bf16)(acc[i][j][r] * scale);
    }
}

// -------------------- f32 GEMM + bias: out = A @ Bt^T + bias ---------------
__global__ void __launch_bounds__(128, 1)
k_gemm_f32(const bf16* __restrict__ A,
           const bf16* __restrict__ Bt,
           const float* __restrict__ bias,
           float* __restrict__ C,
           int cols, int kd) {
  int wave = (blockIdx.x * blockDim.x + threadIdx.x) >> 5;
  int lane = threadIdx.x & 31;
  int l16 = lane & 15, hf = lane >> 4;
  int tn = cols >> 6;
  int m0 = (wave / tn) << 5;
  int n0 = (wave % tn) << 6;
  const bf16* ar[2];
  ar[0] = A + (size_t)(m0 + l16) * kd;
  ar[1] = ar[0] + (size_t)16 * kd;
  const bf16* bc = Bt + (size_t)(n0 + l16) * kd;
  v8f acc[2][4];
#pragma unroll
  for (int i = 0; i < 2; ++i)
#pragma unroll
    for (int j = 0; j < 4; ++j) acc[i][j] = (v8f){};

  for (int kb = 0; kb < kd; kb += 32) {
    v16bf av[2], bv[4];
#pragma unroll
    for (int i = 0; i < 2; ++i) av[i] = load_a_frag(ar[i], kb, hf);
#pragma unroll
    for (int j = 0; j < 4; ++j) bv[j] = load_b_frag(bc + (size_t)(j * 16) * kd, kb, hf);
#pragma unroll
    for (int i = 0; i < 2; ++i)
#pragma unroll
      for (int j = 0; j < 4; ++j)
        acc[i][j] = wmma_bf16(av[i], bv[j], acc[i][j]);
  }
#pragma unroll
  for (int j = 0; j < 4; ++j) {
    float bb = bias[n0 + j * 16 + l16];
#pragma unroll
    for (int i = 0; i < 2; ++i) {
      float* crow = C + (size_t)(m0 + i * 16 + hf * 8) * cols + n0 + j * 16 + l16;
#pragma unroll
      for (int r = 0; r < 8; ++r) crow[(size_t)r * cols] = acc[i][j][r] + bb;
    }
  }
}

// -------------------- flash attention ---------------------------------------
// One wave per (b, h, 16-row M tile). K/V chunks staged into LDS — via the
// gfx1250 async global->LDS path when the builtin exists (ASYNCcnt +
// s_wait_asynccnt), else via registers. QK^T: 4 WMMAs; online softmax;
// P through LDS (C-layout -> A-layout); PV: 4 independent WMMAs.
__global__ void __launch_bounds__(32, 1)
k_attn(const bf16* __restrict__ Qm,
       const bf16* __restrict__ Km,
       const bf16* __restrict__ Vm,
       bf16* __restrict__ Om) {
  __shared__ __align__(16) bf16 sK[32 * 64];  // [n_local][d]
  __shared__ __align__(16) bf16 sV[32 * 64];  // [n_local][d]
  __shared__ __align__(16) bf16 sP[16 * 32];  // [m][n_local]

  int id = blockIdx.x;             // b*256 + h*32 + mt
  int mt = id & 31;
  int h  = (id >> 5) & 7;
  int b  = id >> 8;
  int lane = threadIdx.x;
  int l16 = lane & 15, hf = lane >> 4;
  int m0 = mt << 4;

  const bf16* qrow = Qm + ((size_t)(b * kM + m0 + l16)) * kINNER + h * kDH;
  v16bf aq0 = load_a_frag(qrow, 0, hf);
  v16bf aq1 = load_a_frag(qrow, 32, hf);

  float mrun[8], lrun[8];
  v8f oacc[4];
#pragma unroll
  for (int r = 0; r < 8; ++r) { mrun[r] = -1e30f; lrun[r] = 0.f; }
#pragma unroll
  for (int dt = 0; dt < 4; ++dt) oacc[dt] = (v8f){};

  const bf16* kbase = Km + ((size_t)b * kN) * kINNER + h * kDH;
  const bf16* vbase = Vm + ((size_t)b * kN) * kINNER + h * kDH;

  for (int n0 = 0; n0 < kN; n0 += 32) {
    const bf16* krow_g = kbase + (size_t)(n0 + lane) * kINNER;
    const bf16* vrow_g = vbase + (size_t)(n0 + lane) * kINNER;
#if defined(HAVE_ASYNC_LDS)
    {
#pragma unroll
      for (int c = 0; c < 8; ++c) {
        __builtin_amdgcn_global_load_async_to_lds_b128(
            (GLOBAL_AS v4i*)(krow_g + c * 8),
            (LDS_AS v4i*)(sK + lane * 64 + c * 8), 0, 0);
        __builtin_amdgcn_global_load_async_to_lds_b128(
            (GLOBAL_AS v4i*)(vrow_g + c * 8),
            (LDS_AS v4i*)(sV + lane * 64 + c * 8), 0, 0);
      }
    }
    asm volatile("s_wait_asynccnt 0" ::: "memory");
#else
    {
      const v8bf* kr = (const v8bf*)krow_g;
      const v8bf* vr = (const v8bf*)vrow_g;
      v8bf* skrow = (v8bf*)(sK + lane * 64);
      v8bf* svrow = (v8bf*)(sV + lane * 64);
#pragma unroll
      for (int c = 0; c < 8; ++c) { skrow[c] = kr[c]; svrow[c] = vr[c]; }
    }
#endif
    if (n0 + 32 < kN) {
      __builtin_prefetch(kbase + (size_t)(n0 + 32 + lane) * kINNER, 0, 1);
      __builtin_prefetch(vbase + (size_t)(n0 + 32 + lane) * kINNER, 0, 1);
    }
    __builtin_amdgcn_wave_barrier();   // single wave: HW DS is in-order

    // ---- S = Q K^T : two 16x16 tiles (n sub-tiles), K-dim = 64 ----
    v8f st[2];
#pragma unroll
    for (int sub = 0; sub < 2; ++sub) {
      const bf16* krow = sK + (sub * 16 + l16) * 64;
      v16bf bk0 = load_b_frag(krow, 0, hf);
      v16bf bk1 = load_b_frag(krow, 32, hf);
      v8f s = {};
      s = wmma_bf16(aq0, bk0, s);
      s = wmma_bf16(aq1, bk1, s);
      st[sub] = s;
    }

    // ---- online softmax (row M = r + 8*hf; stats live in 16-lane groups) --
#pragma unroll
    for (int r = 0; r < 8; ++r) {
      float mx = fmaxf(st[0][r], st[1][r]);
#pragma unroll
      for (int m = 1; m <= 8; m <<= 1) mx = fmaxf(mx, __shfl_xor(mx, m, 32));
      float mnew = fmaxf(mrun[r], mx);
      float sc = __expf(mrun[r] - mnew);
      float p0 = __expf(st[0][r] - mnew);
      float p1 = __expf(st[1][r] - mnew);
      float rs = p0 + p1;
#pragma unroll
      for (int m = 1; m <= 8; m <<= 1) rs += __shfl_xor(rs, m, 32);
      lrun[r] = lrun[r] * sc + rs;
      mrun[r] = mnew;
#pragma unroll
      for (int dt = 0; dt < 4; ++dt) oacc[dt][r] *= sc;
      sP[(r + hf * 8) * 32 + l16]      = (bf16)p0;
      sP[(r + hf * 8) * 32 + 16 + l16] = (bf16)p1;
    }
    __builtin_amdgcn_wave_barrier();

    // ---- P as A-operand (16x32); V B-frags gathered from [n][d] LDS ----
    v16bf ap;
    {
      v8bf p0 = *(const v8bf*)(sP + l16 * 32 + hf * 8);
      v8bf p1 = *(const v8bf*)(sP + l16 * 32 + 16 + hf * 8);
#pragma unroll
      for (int t = 0; t < 8; ++t) { ap[t] = p0[t]; ap[t + 8] = p1[t]; }
    }
#pragma unroll
    for (int dt = 0; dt < 4; ++dt) {
      v16bf bv;
#pragma unroll
      for (int e = 0; e < 16; ++e)
        bv[e] = sV[((e & 8) * 2 + hf * 8 + (e & 7)) * 64 + dt * 16 + l16];
      oacc[dt] = wmma_bf16(ap, bv, oacc[dt]);
    }
    __builtin_amdgcn_wave_barrier();
  }

  // ---- normalize and store (row M = r + 8*hf, col = h*64 + dt*16 + l16) ----
  bf16* orow = Om + ((size_t)(b * kM + m0 + hf * 8)) * kINNER + h * kDH + l16;
#pragma unroll
  for (int r = 0; r < 8; ++r) {
    float inv = 1.0f / lrun[r];
#pragma unroll
    for (int dt = 0; dt < 4; ++dt)
      orow[(size_t)r * kINNER + dt * 16] = (bf16)(oacc[dt][r] * inv);
  }
}

// ---------------------------------------------------------------------------
extern "C" void kernel_launch(void* const* d_in, const int* in_sizes, int n_in,
                              void* d_out, int out_size, void* d_ws,
                              size_t ws_size, hipStream_t stream) {
  (void)in_sizes; (void)n_in; (void)out_size; (void)ws_size;
  const float* x   = (const float*)d_in[0];
  const float* lat = (const float*)d_in[1];
  const float* Wq  = (const float*)d_in[2];
  const float* Wk  = (const float*)d_in[3];
  const float* Wv  = (const float*)d_in[4];
  const float* Wo  = (const float*)d_in[5];
  const float* bo  = (const float*)d_in[6];
  const float* gx  = (const float*)d_in[7];
  const float* bx  = (const float*)d_in[8];
  const float* gl  = (const float*)d_in[9];
  const float* bl  = (const float*)d_in[10];
  float* out = (float*)d_out;

  char* ws = (char*)d_ws;
  bf16* xn  = (bf16*)ws; ws += (size_t)kB * kN * kDIM * 2;      // 32 MB
  bf16* lnl = (bf16*)ws; ws += (size_t)kB * kM * kDIM * 2;      //  4 MB
  bf16* WqT = (bf16*)ws; ws += (size_t)kDIM * kINNER * 2;       //  1 MB
  bf16* WkT = (bf16*)ws; ws += (size_t)kDIM * kINNER * 2;
  bf16* WvT = (bf16*)ws; ws += (size_t)kDIM * kINNER * 2;
  bf16* WoT = (bf16*)ws; ws += (size_t)kDIM * kINNER * 2;
  bf16* qb  = (bf16*)ws; ws += (size_t)kB * kM * kINNER * 2;    //  2 MB
  bf16* kb_ = (bf16*)ws; ws += (size_t)kB * kN * kINNER * 2;    // 16 MB
  bf16* vb_ = (bf16*)ws; ws += (size_t)kB * kN * kINNER * 2;    // 16 MB
  bf16* ao  = (bf16*)ws; ws += (size_t)kB * kM * kINNER * 2;    //  2 MB

  // 1) weights -> transposed bf16
  int wE = kDIM * kINNER;
  k_cvt_t<<<(wE + 255) / 256, 256, 0, stream>>>(Wq, WqT, kDIM, kINNER);
  k_cvt_t<<<(wE + 255) / 256, 256, 0, stream>>>(Wk, WkT, kDIM, kINNER);
  k_cvt_t<<<(wE + 255) / 256, 256, 0, stream>>>(Wv, WvT, kDIM, kINNER);
  k_cvt_t<<<(wE + 255) / 256, 256, 0, stream>>>(Wo, WoT, kINNER, kDIM);

  // 2) layernorms -> bf16 activations
  k_ln<<<(kB * kN) / 8, 256, 0, stream>>>(x,   gx, bx, xn,  kB * kN);
  k_ln<<<(kB * kM) / 8, 256, 0, stream>>>(lat, gl, bl, lnl, kB * kM);

  // 3) projections (q with SCALE folded in); 32x64 tiles per wave
  int tq  = ((kB * kM) / 32) * (kINNER / 64);   // 512 waves
  int tkv = ((kB * kN) / 32) * (kINNER / 64);   // 4096 waves
  k_gemm_bf16<<<tq  / 4, 128, 0, stream>>>(lnl, WqT, qb,  kINNER, kDIM, kSCALE);
  k_gemm_bf16<<<tkv / 4, 128, 0, stream>>>(xn,  WkT, kb_, kINNER, kDIM, 1.0f);
  k_gemm_bf16<<<tkv / 4, 128, 0, stream>>>(xn,  WvT, vb_, kINNER, kDIM, 1.0f);

  // 4) flash attention: one wave per (b, h, 16-row tile of M)
  k_attn<<<kB * kHEADS * (kM / 16), 32, 0, stream>>>(qb, kb_, vb_, ao);

  // 5) output projection + bias (f32 out)
  int to = ((kB * kM) / 32) * (kDIM / 64);      // 1024 waves
  k_gemm_f32<<<to / 4, 128, 0, stream>>>(ao, WoT, bo, out, kDIM, kINNER);
}
